// RoutedHybridBlock_40063454937776
// MI455X (gfx1250) — compile-verified, matrix operand order checked by reading, not run on hardware
//
#include <hip/hip_runtime.h>
#include <hip/hip_bf16.h>

// ---------------- problem constants (match reference) ----------------
#define B_    4
#define T_    1024
#define D_    1024
#define H_    16
#define DS_   16
#define DC_   4
#define DI_   2048      // EXP*D
#define DTR_  64        // D/16
#define KTOK_ 204       // int(0.2*T)
#define HD_   64        // D/H

typedef __bf16 bf16_t;
typedef __attribute__((ext_vector_type(16))) __bf16 v16bf;
typedef __attribute__((ext_vector_type(8)))  __bf16 v8bf;
typedef __attribute__((ext_vector_type(8)))  float  v8f;

__device__ __forceinline__ float silu_f(float x) { return x / (1.f + __expf(-x)); }

// Pack a 16-element bf16 WMMA operand fragment from two contiguous 16B runs:
// elements 0..7 = p[0..7] (VGPRs 0-3), elements 8..15 = p[16..23] (VGPRs 4-7),
// matching the ISA 16-bit A/B layout (per-lane K split kb / kb+16).
__device__ __forceinline__ v16bf pack_frag(const bf16_t* p) {
  v8bf lo = *(const v8bf*)(p);
  v8bf hi = *(const v8bf*)(p + 16);
  return __builtin_shufflevector(lo, hi, 0,1,2,3,4,5,6,7,8,9,10,11,12,13,14,15);
}

// =====================================================================
// Generic bf16 WMMA GEMM:  C[M,N] = A[M,K] * B[K,N]  (B given transposed
// as BT[N,K] row-major so per-lane K runs are contiguous 16B loads).
// mode: 0 = none, 1 = +bias, 2 = +bias then softplus.
// Optional bf16 mirror of C written to Cbf (ldcb) when non-null.
// Register blocking: each wave computes a 32x32 output (2x2 16x16 tiles,
// 4 accumulators) so each A/B fragment feeds two WMMAs (2x arithmetic
// intensity vs 1 tile/wave). Block = 256 threads = 8 waves arranged
// 2(M) x 4(N) -> 64 x 128 block tile.
// =====================================================================
__global__ __launch_bounds__(256) void wmma_gemm_kernel(
    const bf16_t* __restrict__ A, int lda,
    const bf16_t* __restrict__ BT,
    const float*  __restrict__ bias,
    float* __restrict__ C, int ldc,
    bf16_t* __restrict__ Cbf, int ldcb,
    int M, int N, int Kdim, int mode)
{
  const int wave = threadIdx.x >> 5;
  const int lane = threadIdx.x & 31;
  const int m0 = blockIdx.y * 64  + (wave >> 2) * 32;
  const int n0 = blockIdx.x * 128 + (wave & 3) * 32;
  const int half = lane >> 4;          // 0: K 0..7/16..23   1: K 8..15/24..31
  const int kb   = half * 8;
  const int l15  = lane & 15;
  const int ar0 = m0 + l15,      ar1 = m0 + 16 + l15;   // A rows (2 M-tiles)
  const int bc0 = n0 + l15,      bc1 = n0 + 16 + l15;   // B cols (2 N-tiles)
  const bf16_t* ap0 = A  + (size_t)(ar0 < M ? ar0 : 0) * (size_t)lda  + kb;
  const bf16_t* ap1 = A  + (size_t)(ar1 < M ? ar1 : 0) * (size_t)lda  + kb;
  const bf16_t* bp0 = BT + (size_t)(bc0 < N ? bc0 : 0) * (size_t)Kdim + kb;
  const bf16_t* bp1 = BT + (size_t)(bc1 < N ? bc1 : 0) * (size_t)Kdim + kb;

  v8f acc00 = {}, acc01 = {}, acc10 = {}, acc11 = {};
  for (int k0 = 0; k0 < Kdim; k0 += 32) {
    const v16bf a0 = pack_frag(ap0 + k0);
    const v16bf a1 = pack_frag(ap1 + k0);
    const v16bf b0 = pack_frag(bp0 + k0);
    const v16bf b1 = pack_frag(bp1 + k0);
    acc00 = __builtin_amdgcn_wmma_f32_16x16x32_bf16(false, a0, false, b0, (short)0, acc00, false, false);
    acc01 = __builtin_amdgcn_wmma_f32_16x16x32_bf16(false, a0, false, b1, (short)0, acc01, false, false);
    acc10 = __builtin_amdgcn_wmma_f32_16x16x32_bf16(false, a1, false, b0, (short)0, acc10, false, false);
    acc11 = __builtin_amdgcn_wmma_f32_16x16x32_bf16(false, a1, false, b1, (short)0, acc11, false, false);
  }

  // D layout per tile: VGPR i holds row tile_m0 + half*8 + i, col tile_n0 + l15
#pragma unroll
  for (int ti = 0; ti < 2; ++ti) {
#pragma unroll
    for (int tj = 0; tj < 2; ++tj) {
      const v8f a = (ti == 0) ? (tj == 0 ? acc00 : acc01)
                              : (tj == 0 ? acc10 : acc11);
      const int col = n0 + tj * 16 + l15;
      const int mb  = m0 + ti * 16 + half * 8;
      if (col < N) {
        const float bv_ = (mode > 0) ? bias[col] : 0.f;
#pragma unroll
        for (int i = 0; i < 8; ++i) {
          const int row = mb + i;
          if (row < M) {
            float v = a[i] + bv_;
            if (mode == 2) v = (v > 20.f) ? v : log1pf(__expf(v));   // softplus
            C[(size_t)row * ldc + col] = v;
            if (Cbf) Cbf[(size_t)row * ldcb + col] = (bf16_t)v;
          }
        }
      }
    }
  }
}

// ---------------- f32 -> bf16 transposed weight conversion ----------------
__global__ void convertT_kernel(const float* __restrict__ W, bf16_t* __restrict__ WT,
                                int Kd, int Nd)
{
  size_t gid = (size_t)blockIdx.x * 256 + threadIdx.x;
  if (gid >= (size_t)Kd * Nd) return;
  int n = (int)(gid % Nd);
  int k = (int)(gid / Nd);
  WT[(size_t)n * Kd + k] = (bf16_t)W[gid];
}

// ---------------- LayerNorm (row per block) ----------------
__global__ __launch_bounds__(256) void layernorm_kernel(
    const float* __restrict__ x, const float* __restrict__ g, const float* __restrict__ be,
    float* __restrict__ xn, bf16_t* __restrict__ xn_bf)
{
  __shared__ float r1[256], r2[256];
  const int row = blockIdx.x;
  const float* xr = x + (size_t)row * D_;
  float v[4], s = 0.f, ss = 0.f;
#pragma unroll
  for (int i = 0; i < 4; ++i) {
    v[i] = xr[threadIdx.x + i * 256];
    s += v[i]; ss += v[i] * v[i];
  }
  r1[threadIdx.x] = s; r2[threadIdx.x] = ss;
  __syncthreads();
  for (int st = 128; st > 0; st >>= 1) {
    if (threadIdx.x < st) { r1[threadIdx.x] += r1[threadIdx.x + st];
                            r2[threadIdx.x] += r2[threadIdx.x + st]; }
    __syncthreads();
  }
  const float mean = r1[0] * (1.f / D_);
  const float var  = r2[0] * (1.f / D_) - mean * mean;
  const float rstd = rsqrtf(var + 1e-5f);
#pragma unroll
  for (int i = 0; i < 4; ++i) {
    int c = threadIdx.x + i * 256;
    float o = (v[i] - mean) * rstd * g[c] + be[c];
    xn[(size_t)row * D_ + c] = o;
    xn_bf[(size_t)row * D_ + c] = (bf16_t)o;
  }
}

// ---------------- Router MLP: relu(xn@Wr1+br1)@Wr2+br2 ----------------
__global__ __launch_bounds__(256) void router_kernel(
    const float* __restrict__ xn, const float* __restrict__ Wr1, const float* __restrict__ br1,
    const float* __restrict__ Wr2, const float* __restrict__ br2, float* __restrict__ scores)
{
  __shared__ float red[256];
  const int row = blockIdx.x;                 // b*T + t
  const int j   = threadIdx.x;                // hidden unit 0..255
  const float* xr = xn + (size_t)row * D_;
  float acc = br1[j];
  for (int k = 0; k < D_; ++k) acc += xr[k] * Wr1[(size_t)k * 256 + j];
  acc = acc > 0.f ? acc : 0.f;
  red[j] = acc * Wr2[j];
  __syncthreads();
  for (int st = 128; st > 0; st >>= 1) {
    if (j < st) red[j] += red[j + st];
    __syncthreads();
  }
  if (j == 0) scores[row] = red[0] + br2[0];
}

// ---------------- top-K by rank counting (matches jax tie-break) ----------------
__global__ __launch_bounds__(256) void topk_kernel(
    const float* __restrict__ scores, int* __restrict__ idx, int* __restrict__ slot)
{
  __shared__ float sc[T_];
  const int b = blockIdx.x;
  for (int i = threadIdx.x; i < T_; i += 256) sc[i] = scores[(size_t)b * T_ + i];
  __syncthreads();
  for (int i = threadIdx.x; i < T_; i += 256) {
    const float si = sc[i];
    int rank = 0;
    for (int j = 0; j < T_; ++j) {
      const float sj = sc[j];
      rank += (sj > si) || (sj == si && j < i);
    }
    if (rank < KTOK_) { idx[b * KTOK_ + rank] = i; slot[(size_t)b * T_ + i] = rank; }
    else              { slot[(size_t)b * T_ + i] = -1; }
  }
}

// ---------------- depthwise causal conv (DC=4) + SiLU ----------------
__global__ void conv_silu_kernel(const float* __restrict__ uz,
                                 const float* __restrict__ conv_w,
                                 const float* __restrict__ conv_b,
                                 float* __restrict__ u_act, bf16_t* __restrict__ u_bf)
{
  size_t gid = (size_t)blockIdx.x * 256 + threadIdx.x;
  if (gid >= (size_t)B_ * T_ * DI_) return;
  const int c  = (int)(gid % DI_);
  const size_t bt = gid / DI_;
  const int t = (int)(bt % T_);
  const size_t brow = bt - t;                  // b*T
  float acc = conv_b[c];
#pragma unroll
  for (int j = 0; j < DC_; ++j) {
    const int tt = t - (DC_ - 1) + j;
    if (tt >= 0) acc += uz[(brow + tt) * (size_t)(2 * DI_) + c] * conv_w[c * DC_ + j];
  }
  const float s = silu_f(acc);
  u_act[gid] = s;
  u_bf[gid]  = (bf16_t)s;
}

// ---------------- selective-scan: h = exp(dt*A)*h + dt*B*u ; y = <h,C> ----------------
// fused with skip + SiLU(z) gate -> y_bf (input to W_out GEMM)
__global__ __launch_bounds__(256) void ssm_scan_kernel(
    const float* __restrict__ proj,   // [B*T, 96], cols 64..79=B, 80..95=C
    const float* __restrict__ dt,     // [B*T, DI]
    const float* __restrict__ u_act,  // [B*T, DI]
    const float* __restrict__ uz,     // [B*T, 2*DI]  (z in cols DI..)
    const float* __restrict__ A_log,  // [DI, DS]
    const float* __restrict__ Dskip,  // [DI]
    bf16_t* __restrict__ y_bf)        // [B*T, DI]
{
  __shared__ float Bs[DS_], Cs[DS_];
  const int b  = blockIdx.x >> 3;                         // DI_/256 = 8 blocks/batch
  const int di = ((blockIdx.x & 7) << 8) + threadIdx.x;
  float A[DS_], h[DS_];
#pragma unroll
  for (int s = 0; s < DS_; ++s) { A[s] = -__expf(A_log[di * DS_ + s]); h[s] = 0.f; }
  const float dsk = Dskip[di];

  for (int t = 0; t < T_; ++t) {
    const size_t r = (size_t)b * T_ + t;
    __syncthreads();
    if (threadIdx.x < 2 * DS_) {
      const float v = proj[r * 96 + DTR_ + threadIdx.x];
      if (threadIdx.x < DS_) Bs[threadIdx.x] = v; else Cs[threadIdx.x - DS_] = v;
    }
    __syncthreads();
    const float dtv = dt[r * DI_ + di];
    const float uv  = u_act[r * DI_ + di];
    float y = 0.f;
#pragma unroll
    for (int s = 0; s < DS_; ++s) {
      h[s] = __expf(dtv * A[s]) * h[s] + dtv * Bs[s] * uv;
      y += h[s] * Cs[s];
    }
    const float z = uz[r * (size_t)(2 * DI_) + DI_ + di];
    y_bf[r * DI_ + di] = (bf16_t)((y + uv * dsk) * silu_f(z));
  }
}

// ---------------- gather routed tokens (bf16) ----------------
__global__ void gather_tok_kernel(const bf16_t* __restrict__ xn_bf,
                                  const int* __restrict__ idx, bf16_t* __restrict__ tok_bf)
{
  size_t gid = (size_t)blockIdx.x * 256 + threadIdx.x;
  if (gid >= (size_t)B_ * KTOK_ * D_) return;
  const int d = (int)(gid % D_);
  const int r = (int)(gid / D_);
  const int b = r / KTOK_, i = r % KTOK_;
  const int t = idx[b * KTOK_ + i];
  tok_bf[gid] = xn_bf[((size_t)b * T_ + t) * D_ + d];
}

// ---------------- attention on routed tokens: one (b,h,query) per block ----------------
__global__ __launch_bounds__(64) void attn_kernel(const float* __restrict__ qkv,
                                                  bf16_t* __restrict__ o_bf)
{
  __shared__ float q_s[HD_], sc[KTOK_], red[64];
  const int qi = blockIdx.x, h = blockIdx.y, b = blockIdx.z;
  const int tid = threadIdx.x;
  const size_t base = (size_t)b * KTOK_;
  q_s[tid] = qkv[(base + qi) * 3072 + h * HD_ + tid];
  __syncthreads();

  float lmax = -3.0e38f;
  for (int j = tid; j < KTOK_; j += 64) {
    const float* kr = qkv + (base + j) * 3072 + D_ + h * HD_;
    float acc = 0.f;
#pragma unroll 8
    for (int d = 0; d < HD_; ++d) acc += q_s[d] * kr[d];
    acc *= 0.125f;                      // 1/sqrt(64)
    sc[j] = acc;
    lmax = fmaxf(lmax, acc);
  }
  red[tid] = lmax; __syncthreads();
  for (int st = 32; st > 0; st >>= 1) {
    if (tid < st) red[tid] = fmaxf(red[tid], red[tid + st]);
    __syncthreads();
  }
  const float mx = red[0];
  __syncthreads();
  float lsum = 0.f;
  for (int j = tid; j < KTOK_; j += 64) { float p = __expf(sc[j] - mx); sc[j] = p; lsum += p; }
  red[tid] = lsum; __syncthreads();
  for (int st = 32; st > 0; st >>= 1) {
    if (tid < st) red[tid] += red[tid + st];
    __syncthreads();
  }
  const float inv = 1.f / red[0];
  float acc = 0.f;
  for (int j = 0; j < KTOK_; ++j)
    acc += sc[j] * qkv[(base + j) * 3072 + 2 * D_ + h * HD_ + tid];
  o_bf[(base + qi) * D_ + h * HD_ + tid] = (bf16_t)(acc * inv);
}

// ---------------- final combine: x + where(mask, attn, ssm) ----------------
__global__ void combine_kernel(const float* __restrict__ x, const int* __restrict__ slot,
                               const float* __restrict__ attn_out,
                               const float* __restrict__ ssm_out, float* __restrict__ out)
{
  size_t gid = (size_t)blockIdx.x * 256 + threadIdx.x;
  if (gid >= (size_t)B_ * T_ * D_) return;
  const int d = (int)(gid % D_);
  const size_t bt = gid / D_;
  const int b = (int)(bt / T_);
  const int sl = slot[bt];
  const float delta = (sl >= 0) ? attn_out[((size_t)b * KTOK_ + sl) * D_ + d]
                                : ssm_out[bt * D_ + d];
  out[gid] = x[gid] + delta;
}

// =====================================================================
extern "C" void kernel_launch(void* const* d_in, const int* in_sizes, int n_in,
                              void* d_out, int out_size, void* d_ws, size_t ws_size,
                              hipStream_t stream) {
  (void)in_sizes; (void)n_in; (void)out_size; (void)ws_size;
  const float* x      = (const float*)d_in[0];
  const float* ln_g   = (const float*)d_in[1];
  const float* ln_b   = (const float*)d_in[2];
  const float* Wr1    = (const float*)d_in[3];
  const float* br1    = (const float*)d_in[4];
  const float* Wr2    = (const float*)d_in[5];
  const float* br2    = (const float*)d_in[6];
  const float* Wqkv   = (const float*)d_in[7];
  const float* bqkv   = (const float*)d_in[8];
  const float* Wo     = (const float*)d_in[9];
  const float* bo     = (const float*)d_in[10];
  const float* W_in   = (const float*)d_in[11];
  const float* conv_w = (const float*)d_in[12];
  const float* conv_b = (const float*)d_in[13];
  const float* W_xprj = (const float*)d_in[14];
  const float* W_dt   = (const float*)d_in[15];
  const float* b_dt   = (const float*)d_in[16];
  const float* A_log  = (const float*)d_in[17];
  const float* Dskip  = (const float*)d_in[18];
  const float* W_out  = (const float*)d_in[19];
  float* out = (float*)d_out;

  // ---- workspace layout (256B aligned slices) ----
  char* ws = (char*)d_ws;
  size_t off = 0;
  auto alloc = [&](size_t bytes) -> char* {
    char* p = ws + off; off += (bytes + 255) & ~(size_t)255; return p;
  };
  const size_t MT = (size_t)B_ * T_;          // 4096 rows
  const size_t MK = (size_t)B_ * KTOK_;       // 816 routed rows
  float*  xn      = (float*) alloc(MT * D_ * 4);
  bf16_t* xn_bf   = (bf16_t*)alloc(MT * D_ * 2);
  bf16_t* WinT    = (bf16_t*)alloc((size_t)4096 * 1024 * 2);
  bf16_t* WxprjT  = (bf16_t*)alloc((size_t)96 * 2048 * 2);
  bf16_t* WdtT    = (bf16_t*)alloc((size_t)2048 * 64 * 2);
  bf16_t* WqkvT   = (bf16_t*)alloc((size_t)3072 * 1024 * 2);
  bf16_t* WoT     = (bf16_t*)alloc((size_t)1024 * 1024 * 2);
  bf16_t* WoutT   = (bf16_t*)alloc((size_t)1024 * 2048 * 2);
  float*  uz      = (float*) alloc(MT * 2 * DI_ * 4);
  float*  u_act   = (float*) alloc(MT * DI_ * 4);
  bf16_t* u_bf    = (bf16_t*)alloc(MT * DI_ * 2);
  float*  proj    = (float*) alloc(MT * 96 * 4);
  bf16_t* proj_bf = (bf16_t*)alloc(MT * 96 * 2);
  float*  dt      = (float*) alloc(MT * DI_ * 4);
  bf16_t* y_bf    = (bf16_t*)alloc(MT * DI_ * 2);
  float*  ssm_out = (float*) alloc(MT * D_ * 4);
  float*  scores  = (float*) alloc(MT * 4);
  int*    idx     = (int*)   alloc((size_t)B_ * KTOK_ * 4);
  int*    slot    = (int*)   alloc(MT * 4);
  bf16_t* tok_bf  = (bf16_t*)alloc(MK * D_ * 2);
  float*  qkv     = (float*) alloc(MK * 3072 * 4);
  bf16_t* o_bf    = (bf16_t*)alloc(MK * D_ * 2);
  float*  attn_o  = (float*) alloc(MK * D_ * 4);

  auto eb = [](size_t n) { return (unsigned)((n + 255) / 256); };   // elementwise blocks
  auto gg = [](int M, int N) { return dim3((N + 127) / 128, (M + 63) / 64, 1); };

  // 1) LayerNorm (+bf16 mirror)
  layernorm_kernel<<<(unsigned)MT, 256, 0, stream>>>(x, ln_g, ln_b, xn, xn_bf);

  // 2) weight transposes -> bf16 [N,K]
  convertT_kernel<<<eb((size_t)1024*4096), 256, 0, stream>>>(W_in,   WinT,   1024, 4096);
  convertT_kernel<<<eb((size_t)2048*96),   256, 0, stream>>>(W_xprj, WxprjT, 2048, 96);
  convertT_kernel<<<eb((size_t)64*2048),   256, 0, stream>>>(W_dt,   WdtT,   64,   2048);
  convertT_kernel<<<eb((size_t)1024*3072), 256, 0, stream>>>(Wqkv,   WqkvT,  1024, 3072);
  convertT_kernel<<<eb((size_t)1024*1024), 256, 0, stream>>>(Wo,     WoT,    1024, 1024);
  convertT_kernel<<<eb((size_t)2048*1024), 256, 0, stream>>>(W_out,  WoutT,  2048, 1024);

  // 3) router + top-k
  router_kernel<<<(unsigned)MT, 256, 0, stream>>>(xn, Wr1, br1, Wr2, br2, scores);
  topk_kernel<<<B_, 256, 0, stream>>>(scores, idx, slot);

  // 4) SSM branch
  wmma_gemm_kernel<<<gg((int)MT, 2*DI_), 256, 0, stream>>>(      // xn @ W_in -> uz
      xn_bf, D_, WinT, nullptr, uz, 2*DI_, nullptr, 0, (int)MT, 2*DI_, D_, 0);
  conv_silu_kernel<<<eb(MT * DI_), 256, 0, stream>>>(uz, conv_w, conv_b, u_act, u_bf);
  wmma_gemm_kernel<<<gg((int)MT, 96), 256, 0, stream>>>(         // u @ W_xproj -> proj
      u_bf, DI_, WxprjT, nullptr, proj, 96, proj_bf, 96, (int)MT, 96, DI_, 0);
  wmma_gemm_kernel<<<gg((int)MT, DI_), 256, 0, stream>>>(        // softplus(proj[:,:64]@W_dt + b_dt)
      proj_bf, 96, WdtT, b_dt, dt, DI_, nullptr, 0, (int)MT, DI_, DTR_, 2);
  ssm_scan_kernel<<<B_ * (DI_/256), 256, 0, stream>>>(proj, dt, u_act, uz, A_log, Dskip, y_bf);
  wmma_gemm_kernel<<<gg((int)MT, D_), 256, 0, stream>>>(         // y @ W_out -> ssm_out
      y_bf, DI_, WoutT, nullptr, ssm_out, D_, nullptr, 0, (int)MT, D_, DI_, 0);

  // 5) attention branch on routed tokens
  gather_tok_kernel<<<eb(MK * D_), 256, 0, stream>>>(xn_bf, idx, tok_bf);
  wmma_gemm_kernel<<<gg((int)MK, 3072), 256, 0, stream>>>(       // tok @ Wqkv + bqkv
      tok_bf, D_, WqkvT, bqkv, qkv, 3072, nullptr, 0, (int)MK, 3072, D_, 1);
  attn_kernel<<<dim3(KTOK_, H_, B_), 64, 0, stream>>>(qkv, o_bf);
  wmma_gemm_kernel<<<gg((int)MK, D_), 256, 0, stream>>>(         // o @ Wo + bo
      o_bf, D_, WoT, bo, attn_o, D_, nullptr, 0, (int)MK, D_, D_, 1);

  // 6) combine
  combine_kernel<<<eb(MT * D_), 256, 0, stream>>>(x, slot, attn_o, ssm_out, out);
}
